// Attention_15461882265636
// MI455X (gfx1250) — matrix operand order repaired and checked by reading, not compiled, not run on hardware
//
#include <hip/hip_runtime.h>
#include <hip/hip_bf16.h>

// MI455X / gfx1250 attention forward, bf16 WMMA (f32 accum), flash-attention style.
// Register-blocked: GEMM waves compute 64x64 tiles (16 wmma per 16 b128 loads),
// attention waves compute 32 query rows (16 wmma per 16 b128 loads per j-step).

typedef __bf16 bf16_t;
typedef __attribute__((ext_vector_type(16))) __bf16 v16bf;
typedef __attribute__((ext_vector_type(8)))  __bf16 v8bf;
typedef __attribute__((ext_vector_type(8)))  float  v8f;

#define DIMC   1024
#define HEADS  16
#define DHEAD  64
#define BATCH  4
#define SEQ    2048
#define TOKENS (BATCH*SEQ)   // 8192
#define NQKV   (3*DIMC)      // 3072

// ---- workspace layout (bytes) ----
#define WS_XB   0UL                      // 8192*1024 bf16 = 16777216 B
#define WS_WTQ  16777216UL               // 1024*3072 bf16 =  6291456 B
#define WS_Q    23068672UL               // 8388608 bf16   = 16777216 B
#define WS_KT   39845888UL
#define WS_V    56623104UL
#define WS_OB   73400320UL
#define WS_WTP  90177536UL               // 1024*1024 bf16 = 2097152 B

static __device__ __forceinline__ v8f wmma_bf16(v16bf a, v16bf b, v8f c) {
  return __builtin_amdgcn_wmma_f32_16x16x32_bf16(false, a, false, b, (short)0, c,
                                                 false, false);
}

// A operand (16x32 bf16, MxK): lane 0-15 row m=lane, K halves 0..7 / 16..23;
// lanes 16-31 same rows, K halves 8..15 / 24..31. Two 16B segment loads.
static __device__ __forceinline__ v16bf load_a_row(const bf16_t* row, int kc, int hi) {
  const v8bf lo = *(const v8bf*)(row + kc + (hi ? 8 : 0));
  const v8bf hv = *(const v8bf*)(row + kc + 16 + (hi ? 8 : 0));
  v16bf a;
#pragma unroll
  for (int i = 0; i < 8; ++i) { a[i] = lo[i]; a[i + 8] = hv[i]; }
  return a;
}

// B operand (32x16 bf16, KxN): lane l holds column n=l%16; K 8h..8h+7 / 16+8h..16+8h+7 (h=l/16), strided gather.
static __device__ __forceinline__ v16bf load_b16(const bf16_t* p, long stride, int hi) {
  v16bf b;
#pragma unroll
  for (int i = 0; i < 8; ++i) {
    b[i] = p[(long)(8 * hi + i) * stride];
    b[i + 8] = p[(long)(16 + 8 * hi + i) * stride];
  }
  return b;
}

static __device__ __forceinline__ v8f vzero8() {
  v8f z;
#pragma unroll
  for (int i = 0; i < 8; ++i) z[i] = 0.0f;
  return z;
}

// ---------------- elementwise converters ----------------
__global__ void k_cvt_bf16(const float* __restrict__ src, bf16_t* __restrict__ dst, int n) {
  int i = blockIdx.x * blockDim.x + threadIdx.x;
  if (i < n) dst[i] = (bf16_t)src[i];
}

// dst[C x R] = transpose(src[R x C]) with fp32 -> bf16 convert
__global__ void k_transpose_cvt(const float* __restrict__ src, bf16_t* __restrict__ dst,
                                int R, int C) {
  int i = blockIdx.x * blockDim.x + threadIdx.x;
  if (i < R * C) {
    int r = i / C, c = i % C;
    dst[(long)c * R + r] = (bf16_t)src[i];
  }
}

// ---------------- QKV GEMM (64x64 per wave) + scatter ----------------
__global__ void __launch_bounds__(256)
k_gemm_qkv(const bf16_t* __restrict__ xb, const bf16_t* __restrict__ wt,
           bf16_t* __restrict__ Q, bf16_t* __restrict__ Kt, bf16_t* __restrict__ V) {
  const int lane = threadIdx.x & 31;
  const int wave = blockIdx.x * (blockDim.x >> 5) + (threadIdx.x >> 5);
  const int NT = NQKV / 64;              // 48 n-tiles
  const int mt = wave / NT;
  const int nt = wave % NT;
  if (mt >= TOKENS / 64) return;
  const int m0 = mt * 64, n0 = nt * 64;
  const int hi = lane >> 4, lm = lane & 15;

  const bf16_t* arow[4];
#pragma unroll
  for (int rc = 0; rc < 4; ++rc) arow[rc] = xb + (long)(m0 + rc * 16 + lm) * DIMC;

  v8f acc[4][4];
#pragma unroll
  for (int rc = 0; rc < 4; ++rc)
#pragma unroll
    for (int nc = 0; nc < 4; ++nc) acc[rc][nc] = vzero8();

#pragma unroll 2
  for (int kc = 0; kc < DIMC; kc += 32) {
    v16bf a[4], b[4];
#pragma unroll
    for (int rc = 0; rc < 4; ++rc) a[rc] = load_a_row(arow[rc], kc, hi);
    const bf16_t* brow = wt + (long)kc * NQKV + n0 + lm;
#pragma unroll
    for (int nc = 0; nc < 4; ++nc) b[nc] = load_b16(brow + nc * 16, NQKV, hi);
#pragma unroll
    for (int rc = 0; rc < 4; ++rc)
#pragma unroll
      for (int nc = 0; nc < 4; ++nc)
        acc[rc][nc] = wmma_bf16(a[rc], b[nc], acc[rc][nc]);
  }

  // 64-wide n-tile aligns exactly with one (q/k/v, head) block: DHEAD == 64
  const int part = n0 / DIMC;            // 0=q 1=k 2=v
  const int h = (n0 % DIMC) / DHEAD;
  const int b = m0 / SEQ;                // 64 | 2048, fixed per tile
  const int i0 = m0 % SEQ;
  const long bh = (long)b * HEADS + h;
#pragma unroll
  for (int rc = 0; rc < 4; ++rc) {
#pragma unroll
    for (int nc = 0; nc < 4; ++nc) {
      const int d = nc * 16 + lm;
#pragma unroll
      for (int j = 0; j < 8; ++j) {
        const int i = i0 + rc * 16 + j + 8 * hi;  // C/D row m = vgpr + 8*(lane>=16)
        const bf16_t val = (bf16_t)acc[rc][nc][j];
        if (part == 0)      Q[(bh * SEQ + i) * DHEAD + d] = val;
        else if (part == 1) Kt[(bh * DHEAD + d) * SEQ + i] = val;
        else                V[(bh * SEQ + i) * DHEAD + d] = val;
      }
    }
  }
}

// ---------------- flash attention: 1 wave per 32 query rows ----------------
__global__ void __launch_bounds__(256)
k_attn(const bf16_t* __restrict__ Q, const bf16_t* __restrict__ Kt,
       const bf16_t* __restrict__ V, bf16_t* __restrict__ Ob) {
  __shared__ bf16_t lds[8 * 32 * 32];    // per-wave 32x32 P tile
  const int lane = threadIdx.x & 31;
  const int wib = threadIdx.x >> 5;
  const int wave = blockIdx.x * 8 + wib; // 4096 waves total
  const int QT = SEQ / 32;               // 64
  const int bh = wave / QT;
  const int q0 = (wave % QT) * 32;
  const int hi = lane >> 4, lm = lane & 15;
  bf16_t* P = lds + wib * (32 * 32);

  v16bf qa[2][2];
#pragma unroll
  for (int rc = 0; rc < 2; ++rc) {
    const bf16_t* qrow = Q + ((long)bh * SEQ + q0 + rc * 16 + lm) * DHEAD;
    qa[rc][0] = load_a_row(qrow, 0, hi);
    qa[rc][1] = load_a_row(qrow, 32, hi);
  }
  const bf16_t* krow0 = Kt + ((long)bh * DHEAD) * SEQ + lm;          // d = 0.., j = lm
  const bf16_t* krow1 = Kt + ((long)bh * DHEAD + 32) * SEQ + lm;     // d = 32.., j = lm

  v8f o[2][4];
  float mrow[2][8], lrow[2][8];
#pragma unroll
  for (int rc = 0; rc < 2; ++rc) {
#pragma unroll
    for (int c = 0; c < 4; ++c) o[rc][c] = vzero8();
#pragma unroll
    for (int j = 0; j < 8; ++j) { mrow[rc][j] = -3.0e38f; lrow[rc][j] = 0.0f; }
  }
  const float scale = 0.125f;            // D^-0.5, D=64

#pragma unroll 1
  for (int j0 = 0; j0 < SEQ; j0 += 32) {
    // K tiles shared across both query row-chunks
    const v16bf kb00 = load_b16(krow0 + j0, SEQ, hi);
    const v16bf kb10 = load_b16(krow1 + j0, SEQ, hi);
    const v16bf kb01 = load_b16(krow0 + j0 + 16, SEQ, hi);
    const v16bf kb11 = load_b16(krow1 + j0 + 16, SEQ, hi);

#pragma unroll
    for (int rc = 0; rc < 2; ++rc) {
      v8f s0 = vzero8(), s1 = vzero8();
      s0 = wmma_bf16(qa[rc][0], kb00, s0);
      s0 = wmma_bf16(qa[rc][1], kb10, s0);
      s1 = wmma_bf16(qa[rc][0], kb01, s1);
      s1 = wmma_bf16(qa[rc][1], kb11, s1);

      // online softmax; row stats reduce across the 16-lane column group
#pragma unroll
      for (int j = 0; j < 8; ++j) {
        float a0 = s0[j] * scale, a1 = s1[j] * scale;
        float mx = fmaxf(a0, a1);
        mx = fmaxf(mx, __shfl_xor(mx, 1, 32));
        mx = fmaxf(mx, __shfl_xor(mx, 2, 32));
        mx = fmaxf(mx, __shfl_xor(mx, 4, 32));
        mx = fmaxf(mx, __shfl_xor(mx, 8, 32));
        const float newm = fmaxf(mrow[rc][j], mx);
        const float corr = __expf(mrow[rc][j] - newm);
        const float p0 = __expf(a0 - newm);
        const float p1 = __expf(a1 - newm);
        float rs = p0 + p1;
        rs += __shfl_xor(rs, 1, 32);
        rs += __shfl_xor(rs, 2, 32);
        rs += __shfl_xor(rs, 4, 32);
        rs += __shfl_xor(rs, 8, 32);
        lrow[rc][j] = lrow[rc][j] * corr + rs;
        mrow[rc][j] = newm;
        o[rc][0][j] *= corr; o[rc][1][j] *= corr;
        o[rc][2][j] *= corr; o[rc][3][j] *= corr;
        s0[j] = p0; s1[j] = p1;
      }

      // C/D layout -> A layout via LDS (wave-private tile; in-wave DScnt dep)
#pragma unroll
      for (int j = 0; j < 8; ++j) {
        const int row = rc * 16 + j + 8 * hi;
        P[row * 32 + lm]      = (bf16_t)s0[j];
        P[row * 32 + 16 + lm] = (bf16_t)s1[j];
      }
    }

    // V tiles shared across both query row-chunks
    const bf16_t* vrow = V + ((long)bh * SEQ + j0) * DHEAD + lm;
    const v16bf vb0 = load_b16(vrow + 0, DHEAD, hi);
    const v16bf vb1 = load_b16(vrow + 16, DHEAD, hi);
    const v16bf vb2 = load_b16(vrow + 32, DHEAD, hi);
    const v16bf vb3 = load_b16(vrow + 48, DHEAD, hi);

#pragma unroll
    for (int rc = 0; rc < 2; ++rc) {
      const v16bf pa = load_a_row(P + (rc * 16 + lm) * 32, 0, hi);
      o[rc][0] = wmma_bf16(pa, vb0, o[rc][0]);
      o[rc][1] = wmma_bf16(pa, vb1, o[rc][1]);
      o[rc][2] = wmma_bf16(pa, vb2, o[rc][2]);
      o[rc][3] = wmma_bf16(pa, vb3, o[rc][3]);
    }
  }

  // normalize and store as [b, i, h*64+d] bf16 for the projection GEMM
  const int b = bh / HEADS, h = bh % HEADS;
#pragma unroll
  for (int rc = 0; rc < 2; ++rc) {
#pragma unroll
    for (int j = 0; j < 8; ++j) {
      const float inv = 1.0f / lrow[rc][j];
      const int i = q0 + rc * 16 + j + 8 * hi;
      const long base = ((long)b * SEQ + i) * DIMC + h * DHEAD;
      Ob[base + lm]      = (bf16_t)(o[rc][0][j] * inv);
      Ob[base + 16 + lm] = (bf16_t)(o[rc][1][j] * inv);
      Ob[base + 32 + lm] = (bf16_t)(o[rc][2][j] * inv);
      Ob[base + 48 + lm] = (bf16_t)(o[rc][3][j] * inv);
    }
  }
}

// ---------------- output projection GEMM (64x64 per wave) + bias ----------------
__global__ void __launch_bounds__(256)
k_gemm_proj(const bf16_t* __restrict__ Ob, const bf16_t* __restrict__ wt,
            const float* __restrict__ bias, float* __restrict__ out) {
  const int lane = threadIdx.x & 31;
  const int wave = blockIdx.x * (blockDim.x >> 5) + (threadIdx.x >> 5);
  const int NT = DIMC / 64;              // 16 n-tiles
  const int mt = wave / NT;
  const int nt = wave % NT;
  if (mt >= TOKENS / 64) return;
  const int m0 = mt * 64, n0 = nt * 64;
  const int hi = lane >> 4, lm = lane & 15;

  const bf16_t* arow[4];
#pragma unroll
  for (int rc = 0; rc < 4; ++rc) arow[rc] = Ob + (long)(m0 + rc * 16 + lm) * DIMC;

  v8f acc[4][4];
#pragma unroll
  for (int rc = 0; rc < 4; ++rc)
#pragma unroll
    for (int nc = 0; nc < 4; ++nc) acc[rc][nc] = vzero8();

#pragma unroll 2
  for (int kc = 0; kc < DIMC; kc += 32) {
    v16bf a[4], b[4];
#pragma unroll
    for (int rc = 0; rc < 4; ++rc) a[rc] = load_a_row(arow[rc], kc, hi);
    const bf16_t* brow = wt + (long)kc * DIMC + n0 + lm;
#pragma unroll
    for (int nc = 0; nc < 4; ++nc) b[nc] = load_b16(brow + nc * 16, DIMC, hi);
#pragma unroll
    for (int rc = 0; rc < 4; ++rc)
#pragma unroll
      for (int nc = 0; nc < 4; ++nc)
        acc[rc][nc] = wmma_bf16(a[rc], b[nc], acc[rc][nc]);
  }

#pragma unroll
  for (int nc = 0; nc < 4; ++nc) {
    const float bb = bias[n0 + nc * 16 + lm];
#pragma unroll
    for (int rc = 0; rc < 4; ++rc) {
#pragma unroll
      for (int j = 0; j < 8; ++j) {
        const int m = m0 + rc * 16 + j + 8 * hi;
        out[(long)m * DIMC + n0 + nc * 16 + lm] = acc[rc][nc][j] + bb;
      }
    }
  }
}

extern "C" void kernel_launch(void* const* d_in, const int* in_sizes, int n_in,
                              void* d_out, int out_size, void* d_ws, size_t ws_size,
                              hipStream_t stream) {
  const float* x     = (const float*)d_in[0];   // [4,2048,1024]
  const float* Wqkv  = (const float*)d_in[1];   // [3072,1024]
  const float* Wproj = (const float*)d_in[2];   // [1024,1024]
  const float* bproj = (const float*)d_in[3];   // [1024]
  float* out = (float*)d_out;

  char* ws = (char*)d_ws;
  bf16_t* xb  = (bf16_t*)(ws + WS_XB);
  bf16_t* wtq = (bf16_t*)(ws + WS_WTQ);
  bf16_t* Q   = (bf16_t*)(ws + WS_Q);
  bf16_t* Kt  = (bf16_t*)(ws + WS_KT);
  bf16_t* V   = (bf16_t*)(ws + WS_V);
  bf16_t* Ob  = (bf16_t*)(ws + WS_OB);
  bf16_t* wtp = (bf16_t*)(ws + WS_WTP);

  // 1) fp32 -> bf16 staging
  k_cvt_bf16<<<(TOKENS * DIMC + 255) / 256, 256, 0, stream>>>(x, xb, TOKENS * DIMC);
  k_transpose_cvt<<<(NQKV * DIMC + 255) / 256, 256, 0, stream>>>(Wqkv, wtq, NQKV, DIMC);
  k_transpose_cvt<<<(DIMC * DIMC + 255) / 256, 256, 0, stream>>>(Wproj, wtp, DIMC, DIMC);

  // 2) QKV GEMM: 128 m-tiles * 48 n-tiles = 6144 waves / 8 per block
  k_gemm_qkv<<<768, 256, 0, stream>>>(xb, wtq, Q, Kt, V);

  // 3) flash attention: 4*16*64 = 4096 waves / 8 per block
  k_attn<<<512, 256, 0, stream>>>(Q, Kt, V, Ob);

  // 4) projection: 128 * 16 = 2048 waves / 8 per block
  k_gemm_proj<<<256, 256, 0, stream>>>(Ob, wtp, bproj, out);
}
